// AdvancedAMLGNN_13546326852034
// MI455X (gfx1250) — matrix operand order repaired and checked by reading, not compiled, not run on hardware
//
#include <hip/hip_runtime.h>
#include <hip/hip_bf16.h>

typedef __attribute__((ext_vector_type(16))) __bf16 v16bf;
typedef __attribute__((ext_vector_type(8)))  __bf16 v8bf;
typedef __attribute__((ext_vector_type(8)))  float  v8f;
typedef __bf16 bf16_t;

#define NHID   256      // H
#define DIN_   128
#define EATTR_ 8
#define BN_EPS 1e-5f
#define LDSPAD 8        // bf16 elements of per-row padding (16B) -> bank rotation
#define USE_ASYNC 1     // gfx1250 GLOBAL_LOAD_ASYNC_TO_LDS_B128 for LDS staging

// ---------------------------------------------------------------------------
// helpers
// ---------------------------------------------------------------------------
struct AF { float v[16]; };   // one wave-lane's raw A fragment (two 8-float runs)

static __device__ __forceinline__ void load_run8(float* r, const float* base) {
  const float4* q = (const float4*)base;      // 32B-aligned (rows are 8-float multiples)
  float4 a = q[0], b = q[1];
  r[0]=a.x; r[1]=a.y; r[2]=a.z; r[3]=a.w;
  r[4]=b.x; r[5]=b.y; r[6]=b.z; r[7]=b.w;
}

static __device__ __forceinline__ AF load_a16(const float* arow, int k0, int kh) {
  AF f;
  load_run8(f.v,     arow + k0 + 16 * kh);      // elems 0..7  : K = k0+16kh..+7
  load_run8(f.v + 8, arow + k0 + 16 * kh + 8);  // elems 8..15 : K = k0+16kh+8..+7
  return f;
}

// virtual edge-feature row: [ hc[src](256) | hc[dst](256) | edge_attr(8) | 0-pad ]
static __device__ __forceinline__ void load8_seg(float* r, const float* hc, const float* ea,
                                                 int srow, int drow, int ge, int k) {
  if (k >= 520) {
#pragma unroll
    for (int i = 0; i < 8; ++i) r[i] = 0.f;
    return;
  }
  const float* p;
  if (k < 256)        p = hc + (size_t)srow * NHID + k;
  else if (k < 512)   p = hc + (size_t)drow * NHID + (k - 256);
  else                p = ea + (size_t)ge * EATTR_ + (k - 512);   // exactly 8 floats
  load_run8(r, p);
}

static __device__ __forceinline__ AF load_a16_edge(const float* hc, const float* ea,
                                                   int sr, int dr, int ge, int k0, int kh) {
  AF f;
  load8_seg(f.v,     hc, ea, sr, dr, ge, k0 + 16 * kh);
  load8_seg(f.v + 8, hc, ea, sr, dr, ge, k0 + 16 * kh + 8);
  return f;
}

static __device__ __forceinline__ v16bf cvt_af(const AF& f) {
  v16bf a;
#pragma unroll
  for (int i = 0; i < 16; ++i) a[i] = (__bf16)f.v[i];
  return a;
}

// stage one 16B chunk of the weight panel into LDS.
// NOTE: ldst is also passed as an (unreferenced) "v" operand so the LDS buffer
// escapes into the asm -- otherwise alias analysis proves `panel` is never
// written and deletes the ds_load reads of the B fragments (observed round 2).
static __device__ __forceinline__ void stage16(bf16_t* ldst, const bf16_t* gsrc,
                                               const bf16_t* lds_base) {
#if USE_ASYNC
  unsigned off = (unsigned)((uintptr_t)ldst - (uintptr_t)lds_base);  // segment offset
  asm volatile("global_load_async_to_lds_b128 %0, %1, off"
               :: "v"(off), "v"(gsrc), "v"(ldst) : "memory");
#else
  *(v8bf*)ldst = *(const v8bf*)gsrc;
#endif
}

static __device__ __forceinline__ void stage_fence() {
#if USE_ASYNC
  asm volatile("s_wait_asynccnt 0x0" ::: "memory");
#endif
  __syncthreads();
}

// read a 32-byte B fragment (16 bf16) from the LDS panel as two 16B loads
static __device__ __forceinline__ v16bf ldsb(const bf16_t* p) {
  v8bf lo = *(const v8bf*)p;
  v8bf hi = *(const v8bf*)(p + 8);
  return __builtin_shufflevector(lo, hi, 0,1,2,3,4,5,6,7,8,9,10,11,12,13,14,15);
}

// ---------------------------------------------------------------------------
// WMMA GEMM: C[M,N] = act(A[M,K](f32) @ B[K,N] + bias)
// B pre-converted & transposed: Bt[n*K + k] (bf16), K % 32 == 0, K <= 256, N % 64 == 0
// block = 256 thr = 8 waves; wave -> 16 rows x 64 cols (4 WMMA tiles)
// weight panel (64 cols x K) staged in LDS (async), A double-buffered in regs
// ---------------------------------------------------------------------------
__global__ void gemm_bf16(const float* __restrict__ A, const bf16_t* __restrict__ Bt,
                          const float* __restrict__ bias, float* __restrict__ C,
                          int M, int N, int K, int relu) {
  __shared__ __attribute__((aligned(16))) bf16_t panel[64 * (256 + LDSPAD)];
  const int Kld = K + LDSPAD;
  const int n0  = blockIdx.y * 64;

  // cooperative panel fill: 16B (8 bf16) chunks, K*8 chunks total
  const int cpr = K >> 3;                    // chunks per column-row
  for (int idx = threadIdx.x; idx < K * 8; idx += 256) {
    int cc = idx / cpr;
    int kk = (idx - cc * cpr) << 3;
    stage16(&panel[cc * Kld + kk], Bt + (size_t)(n0 + cc) * K + kk, panel);
  }
  stage_fence();

  const int lane = threadIdx.x & 31;
  const int wave = threadIdx.x >> 5;
  const int row0 = (blockIdx.x * 8 + wave) * 16;
  const int kh   = lane >> 4;
  const int ml   = lane & 15;

  int m = row0 + ml; if (m > M - 1) m = M - 1;   // clamp tail (stores guarded)
  const float* arow = A + (size_t)m * K;

  v8f acc[4];
#pragma unroll
  for (int t = 0; t < 4; ++t)
#pragma unroll
    for (int i = 0; i < 8; ++i) acc[t][i] = 0.f;

  AF cur = load_a16(arow, 0, kh);                // prologue
  for (int k0 = 0; k0 < K; k0 += 32) {
    AF nxt;
    if (k0 + 32 < K) nxt = load_a16(arow, k0 + 32, kh);   // issue early, hide latency
    const v16bf af = cvt_af(cur);
#pragma unroll
    for (int t = 0; t < 4; ++t) {
      const v16bf bf = ldsb(&panel[(16 * t + ml) * Kld + k0 + 16 * kh]);
      acc[t] = __builtin_amdgcn_wmma_f32_16x16x32_bf16(false, af, false, bf,
                                                       (short)0, acc[t], false, false);
    }
    cur = nxt;
  }
#pragma unroll
  for (int t = 0; t < 4; ++t) {
    const int c = n0 + 16 * t + ml;
    const float bv = bias ? bias[c] : 0.f;
#pragma unroll
    for (int v = 0; v < 8; ++v) {
      const int r = row0 + v + 8 * kh;           // C/D layout: VGPR v -> row v + 8*kh
      if (r < M) {
        float val = acc[t][v] + bv;
        if (relu) val = fmaxf(val, 0.f);
        C[(size_t)r * N + c] = val;
      }
    }
  }
}

// classifier layer 1 with fused gather: rows are edges e0..e0+M-1, K = 544
__global__ void gemm_edge_bf16(const float* __restrict__ hc, const float* __restrict__ eattr,
                               const int* __restrict__ srcI, const int* __restrict__ dstI,
                               const bf16_t* __restrict__ Bt, const float* __restrict__ bias,
                               float* __restrict__ C, int e0, int M, int N, int K, int relu) {
  __shared__ __attribute__((aligned(16))) bf16_t panel[64 * (544 + LDSPAD)];
  const int Kld = K + LDSPAD;
  const int n0  = blockIdx.y * 64;

  const int cpr = K >> 3;
  for (int idx = threadIdx.x; idx < K * 8; idx += 256) {
    int cc = idx / cpr;
    int kk = (idx - cc * cpr) << 3;
    stage16(&panel[cc * Kld + kk], Bt + (size_t)(n0 + cc) * K + kk, panel);
  }
  stage_fence();

  const int lane = threadIdx.x & 31;
  const int wave = threadIdx.x >> 5;
  const int row0 = (blockIdx.x * 8 + wave) * 16;
  const int kh   = lane >> 4;
  const int ml   = lane & 15;

  int m = row0 + ml; if (m > M - 1) m = M - 1;
  const int ge = e0 + m;
  const int sr = srcI[ge], dr = dstI[ge];

  v8f acc[4];
#pragma unroll
  for (int t = 0; t < 4; ++t)
#pragma unroll
    for (int i = 0; i < 8; ++i) acc[t][i] = 0.f;

  AF cur = load_a16_edge(hc, eattr, sr, dr, ge, 0, kh);
  for (int k0 = 0; k0 < K; k0 += 32) {
    AF nxt;
    if (k0 + 32 < K) nxt = load_a16_edge(hc, eattr, sr, dr, ge, k0 + 32, kh);
    const v16bf af = cvt_af(cur);
#pragma unroll
    for (int t = 0; t < 4; ++t) {
      const v16bf bf = ldsb(&panel[(16 * t + ml) * Kld + k0 + 16 * kh]);
      acc[t] = __builtin_amdgcn_wmma_f32_16x16x32_bf16(false, af, false, bf,
                                                       (short)0, acc[t], false, false);
    }
    cur = nxt;
  }
#pragma unroll
  for (int t = 0; t < 4; ++t) {
    const int c = n0 + 16 * t + ml;
    const float bv = bias[c];
#pragma unroll
    for (int v = 0; v < 8; ++v) {
      const int r = row0 + v + 8 * kh;
      if (r < M) {
        float val = acc[t][v] + bv;
        if (relu) val = fmaxf(val, 0.f);
        C[(size_t)r * N + c] = val;
      }
    }
  }
}

// ---------------------------------------------------------------------------
// small kernels
// ---------------------------------------------------------------------------
__global__ void fill_kernel(float* p, float v, size_t n) {
  size_t i = (size_t)blockIdx.x * blockDim.x + threadIdx.x;
  if (i < n) p[i] = v;
}

// Wt[n*Kpad + k] = bf16(W[k*Nout + n]); zero-pad k >= Kreal
__global__ void convw_kernel(const float* __restrict__ W, bf16_t* __restrict__ Wt,
                             int Kreal, int Kpad, int Nout) {
  int idx = blockIdx.x * blockDim.x + threadIdx.x;
  if (idx >= Nout * Kpad) return;
  int n = idx / Kpad, k = idx - n * Kpad;
  Wt[idx] = (k < Kreal) ? (__bf16)W[(size_t)k * Nout + n] : (__bf16)0.f;
}

__global__ void deg_kernel(const int* __restrict__ dstI, float* deg, int E) {
  int e = blockIdx.x * blockDim.x + threadIdx.x;
  if (e < E) atomicAdd(&deg[dstI[e]], 1.f);
}

__global__ void isd_kernel(float* deg, int Nn) {
  int i = blockIdx.x * blockDim.x + threadIdx.x;
  if (i < Nn) deg[i] = rsqrtf(deg[i]);       // deg pre-filled with 1.0 (self loop)
}

// agg = hw * isd^2 + bias   (self-loop term + bias, initializes accumulator)
__global__ void agg_init_kernel(const float* __restrict__ hw, const float* __restrict__ isd,
                                const float* __restrict__ b, float* __restrict__ agg, int Nn) {
  size_t i = (size_t)blockIdx.x * blockDim.x + threadIdx.x;
  if (i >= (size_t)Nn * NHID) return;
  int row = (int)(i >> 8);
  int c = (int)(i & (NHID - 1));
  float s = isd[row];
  agg[i] = hw[i] * s * s + b[c];
}

// agg[dst] += hw[src] * isd[src]*isd[dst]   (8 channels / thread, f32 atomics)
__global__ void scatter_kernel(const float* __restrict__ hw, const float* __restrict__ isd,
                               const int* __restrict__ srcI, const int* __restrict__ dstI,
                               float* __restrict__ agg, int E) {
  int e = blockIdx.x * blockDim.x + threadIdx.x;
  if (e >= E) return;
  int cg = blockIdx.y * 8;
  int s = srcI[e], d = dstI[e];
  float norm = isd[s] * isd[d];
  const float4* hp = (const float4*)(hw + (size_t)s * NHID + cg);
  float4 v0 = hp[0], v1 = hp[1];
  float* ap = agg + (size_t)d * NHID + cg;
  atomicAdd(ap + 0, v0.x * norm); atomicAdd(ap + 1, v0.y * norm);
  atomicAdd(ap + 2, v0.z * norm); atomicAdd(ap + 3, v0.w * norm);
  atomicAdd(ap + 4, v1.x * norm); atomicAdd(ap + 5, v1.y * norm);
  atomicAdd(ap + 6, v1.z * norm); atomicAdd(ap + 7, v1.w * norm);
}

// column sums / sumsq -> stats[0..255] / stats[256..511]
__global__ void bn_stats_kernel(const float* __restrict__ h, float* __restrict__ stats, int Nn) {
  int c = threadIdx.x;                       // 256 channels
  int r0 = blockIdx.x * 256;
  int r1 = r0 + 256; if (r1 > Nn) r1 = Nn;
  float s = 0.f, s2 = 0.f;
  for (int r = r0; r < r1; ++r) {
    float v = h[(size_t)r * NHID + c];
    s += v; s2 += v * v;
  }
  atomicAdd(&stats[c], s);
  atomicAdd(&stats[NHID + c], s2);
}

__global__ void bn_relu_kernel(float* __restrict__ h, const float* __restrict__ stats,
                               const float* __restrict__ g, const float* __restrict__ be, int Nn) {
  size_t i = (size_t)blockIdx.x * blockDim.x + threadIdx.x;
  if (i >= (size_t)Nn * NHID) return;
  int c = (int)(i & (NHID - 1));
  float inv = 1.f / (float)Nn;
  float m = stats[c] * inv;
  float var = stats[NHID + c] * inv - m * m;
  float v = g[c] * (h[i] - m) * rsqrtf(var + BN_EPS) + be[c];
  h[i] = v > 0.f ? v : 0.f;
}

__global__ void copyadd_kernel(const float* __restrict__ src, float* __restrict__ dst,
                               size_t n, int add) {
  size_t i = (size_t)blockIdx.x * blockDim.x + threadIdx.x;
  if (i >= n) return;
  dst[i] = add ? dst[i] + src[i] : src[i];
}

// final 64 -> 2 layer (no relu)
__global__ void mlp5_kernel(const float* __restrict__ Hin, const float* __restrict__ W,
                            const float* __restrict__ b, float* __restrict__ out,
                            int e0, int M) {
  int r = blockIdx.x * blockDim.x + threadIdx.x;
  if (r >= M) return;
  const float* h = Hin + (size_t)r * 64;
  float a0 = b[0], a1 = b[1];
#pragma unroll 4
  for (int k = 0; k < 64; ++k) {
    float v = h[k];
    a0 += v * W[k * 2 + 0];
    a1 += v * W[k * 2 + 1];
  }
  size_t o = (size_t)(e0 + r) * 2;
  out[o] = a0; out[o + 1] = a1;
}

// ---------------------------------------------------------------------------
// host orchestration
// ---------------------------------------------------------------------------
extern "C" void kernel_launch(void* const* d_in, const int* in_sizes, int n_in,
                              void* d_out, int out_size, void* d_ws, size_t ws_size,
                              hipStream_t stream) {
  (void)n_in; (void)out_size;
  const float* x     = (const float*)d_in[0];
  const int*   ei    = (const int*)d_in[1];
  const float* eattr = (const float*)d_in[2];
  const int N = in_sizes[0] / DIN_;
  const int E = in_sizes[1] / 2;
  const int* srcI = ei;
  const int* dstI = ei + E;

  const float *W[2][3], *Bv[2][3];
  for (int br = 0; br < 2; ++br)
    for (int li = 0; li < 3; ++li) {
      W[br][li]  = (const float*)d_in[3 + br * 6 + li * 2];
      Bv[br][li] = (const float*)d_in[4 + br * 6 + li * 2];
    }
  const float *g[3], *be[3];
  for (int li = 0; li < 3; ++li) {
    g[li]  = (const float*)d_in[15 + li * 2];
    be[li] = (const float*)d_in[16 + li * 2];
  }
  const float *Wc[5], *bc[5];
  for (int li = 0; li < 5; ++li) {
    Wc[li] = (const float*)d_in[21 + li * 2];
    bc[li] = (const float*)d_in[22 + li * 2];
  }

  // ---- workspace bump allocator ----
  char* wsp = (char*)d_ws;
  size_t off = 0;
  auto alloc = [&](size_t bytes) -> void* {
    void* p = wsp + off;
    off += (bytes + 255) & ~(size_t)255;
    return p;
  };
  float* isd   = (float*)alloc((size_t)N * 4);
  float* hbuf  = (float*)alloc((size_t)N * NHID * 4);
  float* hw    = (float*)alloc((size_t)N * NHID * 4);
  float* hcb   = (float*)alloc((size_t)N * NHID * 4);
  float* stats = (float*)alloc(2 * NHID * 4);
  const int Kp[3] = {DIN_, NHID, NHID};
  bf16_t* Wt[2][3];
  for (int br = 0; br < 2; ++br)
    for (int li = 0; li < 3; ++li)
      Wt[br][li] = (bf16_t*)alloc((size_t)NHID * Kp[li] * 2);
  const int KcReal[4] = {520, 256, 256, 128};
  const int KcPad [4] = {544, 256, 256, 128};
  const int NcOut [4] = {256, 256, 128,  64};
  bf16_t* Wct[4];
  for (int li = 0; li < 4; ++li)
    Wct[li] = (bf16_t*)alloc((size_t)NcOut[li] * KcPad[li] * 2);

  // edge chunk sized to remaining scratch (two [chunk,256] f32 ping-pong bufs)
  size_t rem = ws_size > off ? ws_size - off - 1024 : 0;
  long long ch = (long long)(rem / ((size_t)2 * NHID * 4));
  if (ch > 100000) ch = 100000;
  if (ch > E) ch = E;
  ch &= ~127LL;
  if (ch < 128) ch = 128;
  const int chunk = (int)ch;
  float* ebufA = (float*)alloc((size_t)chunk * NHID * 4);
  float* ebufB = (float*)alloc((size_t)chunk * NHID * 4);

  auto cdiv = [](long long a, long long b) { return (int)((a + b - 1) / b); };
  auto gemm = [&](const float* A, const bf16_t* Bt, const float* bias, float* C,
                  int M, int Nn, int K, int relu) {
    dim3 grid(cdiv(M, 128), Nn / 64);
    gemm_bf16<<<grid, 256, 0, stream>>>(A, Bt, bias, C, M, Nn, K, relu);
  };

  // ---- weight conversions (f32 [K,N] -> bf16 [N,Kpad] transposed) ----
  for (int br = 0; br < 2; ++br)
    for (int li = 0; li < 3; ++li)
      convw_kernel<<<cdiv((long long)NHID * Kp[li], 256), 256, 0, stream>>>(
          W[br][li], Wt[br][li], Kp[li], Kp[li], NHID);
  for (int li = 0; li < 4; ++li)
    convw_kernel<<<cdiv((long long)NcOut[li] * KcPad[li], 256), 256, 0, stream>>>(
        Wc[li], Wct[li], KcReal[li], KcPad[li], NcOut[li]);

  // ---- degrees: deg = 1 + sum_edges, isd = deg^-1/2 ----
  fill_kernel<<<cdiv(N, 256), 256, 0, stream>>>(isd, 1.f, (size_t)N);
  deg_kernel<<<cdiv(E, 256), 256, 0, stream>>>(dstI, isd, E);
  isd_kernel<<<cdiv(N, 256), 256, 0, stream>>>(isd, N);

  // ---- two GCN branches ----
  const size_t nh = (size_t)N * NHID;
  for (int br = 0; br < 2; ++br) {
    const float* hin = x;
    int kin = DIN_;
    for (int li = 0; li < 3; ++li) {
      gemm(hin, Wt[br][li], nullptr, hw, N, NHID, kin, 0);          // hw = h @ W
      agg_init_kernel<<<cdiv((long long)nh, 256), 256, 0, stream>>>(hw, isd, Bv[br][li], hbuf, N);
      dim3 sg(cdiv(E, 256), NHID / 8);
      scatter_kernel<<<sg, 256, 0, stream>>>(hw, isd, srcI, dstI, hbuf, E);
      fill_kernel<<<cdiv(2 * NHID, 256), 256, 0, stream>>>(stats, 0.f, 2 * NHID);
      bn_stats_kernel<<<cdiv(N, 256), 256, 0, stream>>>(hbuf, stats, N);
      bn_relu_kernel<<<cdiv((long long)nh, 256), 256, 0, stream>>>(hbuf, stats, g[li], be[li], N);
      hin = hbuf; kin = NHID;
    }
    copyadd_kernel<<<cdiv((long long)nh, 256), 256, 0, stream>>>(hbuf, hcb, nh, br); // hc = h1 + h2
  }

  // ---- edge classifier MLP, chunked over edges ----
  for (int e0 = 0; e0 < E; e0 += chunk) {
    int m = E - e0; if (m > chunk) m = chunk;
    dim3 g1(cdiv(m, 128), 4);
    gemm_edge_bf16<<<g1, 256, 0, stream>>>(hcb, eattr, srcI, dstI, Wct[0], bc[0],
                                           ebufA, e0, m, 256, 544, 1);
    gemm(ebufA, Wct[1], bc[1], ebufB, m, 256, 256, 1);
    gemm(ebufB, Wct[2], bc[2], ebufA, m, 128, 256, 1);
    gemm(ebufA, Wct[3], bc[3], ebufB, m,  64, 128, 1);
    mlp5_kernel<<<cdiv(m, 256), 256, 0, stream>>>(ebufB, Wc[4], bc[4], (float*)d_out, e0, m);
  }
}
